// MultiClassAttentionHead_28372553957758
// MI455X (gfx1250) — compile-verified, matrix operand order checked by reading, not run on hardware
//
#include <hip/hip_runtime.h>

#define Bsz 64
#define Ssz 576
#define Dsz 768
#define Csz 200
#define CPAD 224            // 14 n-tiles of 16 (uniform work for all 8 waves)
#define KC 64               // K chunk staged in LDS (two K=32 WMMA steps)
#define KSTR 72             // padded K-stride (bf16 elems): 144B row stride -> conflict-free b128 gathers
#define NKC (Dsz / KC)      // 12 K-chunks

typedef __attribute__((ext_vector_type(8)))  float  v8f;
typedef __attribute__((ext_vector_type(16))) __bf16 v16bf;
typedef unsigned int u32x4 __attribute__((ext_vector_type(4)));
typedef int          i32x4 __attribute__((ext_vector_type(4)));
typedef int          i32x8 __attribute__((ext_vector_type(8)));

#if defined(__HIP_DEVICE_COMPILE__) && __has_builtin(__builtin_amdgcn_tensor_load_to_lds)
#define HAVE_TDM 1
#else
#define HAVE_TDM 0
#endif

union FragU { uint4 q[2]; v16bf v; };

__device__ __forceinline__ unsigned short f2bf_rne(float f) {
  unsigned u = __float_as_uint(f);
  u += 0x7fffu + ((u >> 16) & 1u);
  return (unsigned short)(u >> 16);
}
__device__ __forceinline__ float bf2f(unsigned short h) {
  return __uint_as_float(((unsigned)h) << 16);
}

#if HAVE_TDM
// One contiguous 1-D DMA: global -> LDS, `bytes` must be a multiple of 8.
// D# packing per CDNA5 ISA ch.8: group0 {count, lds_addr, global_addr, type=2},
// group1 {data_size=8B, tensor_dim0 = tile_dim0 = bytes/8, dim1 = 1}.
__device__ __forceinline__ void tdm_load_1d(const void* gptr, unsigned lds_off, unsigned bytes) {
  unsigned long long ga = (unsigned long long)(uintptr_t)gptr;
  unsigned n8 = bytes >> 3;                        // 8-byte elements (fits 16-bit tile_dim0)
  u32x4 g0 = { 1u,                                 // count=1 valid descriptor
               lds_off,                            // LDS byte address
               (unsigned)ga,                       // global_addr[31:0]
               (unsigned)((ga >> 32) & 0x1FFFFFFull) | (2u << 30) };  // ga[56:32] | type=2
  i32x8 g1 = { (int)(3u << 16),                    // data_size = 3 (8 bytes)
               (int)((n8 & 0xFFFFu) << 16),        // tensor_dim0[15:0]
               (int)(((n8 >> 16) & 0xFFFFu) | (1u << 16)),  // tensor_dim0[31:16] | tensor_dim1=1
               (int)((n8 & 0xFFFFu) << 16),        // tile_dim0 = n8
               1,                                  // tile_dim1 = 1, tile_dim2 = 0
               (int)n8,                            // tensor_dim0_stride[31:0]
               0, 0 };
  i32x4 z4 = { 0, 0, 0, 0 };
#if __clang_major__ >= 23
  i32x8 z8 = { 0, 0, 0, 0, 0, 0, 0, 0 };
  __builtin_amdgcn_tensor_load_to_lds(g0, g1, z4, z4, z8, 0);
#else
  __builtin_amdgcn_tensor_load_to_lds(g0, g1, z4, z4, 0);
#endif
}
#endif

// Pre-convert attn_w once into the exact per-chunk LDS image:
// wbf[kc][0=hi|1=lo][row 0..223][k 0..71] bf16, zero padded (rows>=200, k>=64).
__global__ __launch_bounds__(256)
void preconv_w_kernel(const float* __restrict__ w, unsigned short* __restrict__ wbf) {
  int idx = blockIdx.x * 256 + threadIdx.x;
  const int N = NKC * CPAD * KSTR;
  if (idx >= N) return;
  int k   = idx % KSTR;
  int row = (idx / KSTR) % CPAD;
  int kc  = idx / (KSTR * CPAD);
  float x = 0.f;
  if (k < KC && row < Csz) x = w[(size_t)row * Dsz + kc * KC + k];
  unsigned short hi = f2bf_rne(x);
  unsigned short lo = f2bf_rne(x - bf2f(hi));
  wbf[((size_t)(kc * 2 + 0) * CPAD + row) * KSTR + k] = hi;
  wbf[((size_t)(kc * 2 + 1) * CPAD + row) * KSTR + k] = lo;
}

template <bool PRECONV>
__global__ __launch_bounds__(256)
void attn_gemm_kernel(const float* __restrict__ patch,
                      const float* __restrict__ attn_w,
                      const unsigned short* __restrict__ wbf,
                      const float* __restrict__ attn_b,
                      float* __restrict__ partial) {
  __shared__ __attribute__((aligned(16))) unsigned short Ahi[64][KSTR];
  __shared__ __attribute__((aligned(16))) unsigned short Alo[64][KSTR];
  __shared__ __attribute__((aligned(16))) unsigned short Wlds[2][CPAD][KSTR]; // [hi|lo]
  __shared__ float red[256];
  __shared__ float rowsum[64];

  const int tid  = threadIdx.x;
  const int b    = blockIdx.y;
  const int m0   = blockIdx.x * 64;

  const int lane = tid & 31;
  const int w    = tid >> 5;
  const int mrow = w & 3;      // m-subtile (16 rows) of this wave
  const int ncol = w >> 2;     // 0: n-tiles 0..6, 1: n-tiles 7..13
  const int h    = lane >> 4;  // lane half
  const int rl   = lane & 15;  // row (A) / col (B,C) within 16x16 tile

  v8f acc[7] = {};
  float rs_acc = 0.f;

  const int lrow = tid >> 2;   // 0..63: A row this thread loads
  const int lseg = tid & 3;    // 16-float segment within 64-wide K chunk
  const float* aRow = patch + ((size_t)(b * Ssz + m0 + lrow)) * Dsz + lseg * 16;

  for (int kc = 0; kc < NKC; ++kc) {
    const int k0 = kc * KC;
    __syncthreads();   // previous chunk's compute done; LDS reusable

    // ---- kick W chunk staging ----
    if (PRECONV) {
      const unsigned short* src = wbf + (size_t)kc * 2 * CPAD * KSTR;
#if HAVE_TDM
      if (w == 0)   // single wave issues the DMA; overlaps the A-stage below
        tdm_load_1d(src, (unsigned)(uintptr_t)(void*)&Wlds[0][0][0],
                    2u * CPAD * KSTR * sizeof(unsigned short));
#else
      const uint4* s4 = (const uint4*)src;
      uint4* d4 = (uint4*)&Wlds[0][0][0];
      #pragma unroll
      for (int it = 0; it < (2 * CPAD * KSTR) / (8 * 256); ++it)
        d4[tid + it * 256] = s4[tid + it * 256];
      // tail: (2*224*72)/8 = 4032 = 15.75*256
      if (tid < (2 * CPAD * KSTR) / 8 - ((2 * CPAD * KSTR) / (8 * 256)) * 256)
        d4[tid + ((2 * CPAD * KSTR) / (8 * 256)) * 256] =
            s4[tid + ((2 * CPAD * KSTR) / (8 * 256)) * 256];
#endif
    }

    // ---- stage A chunk (64 x 64) as bf16 hi/lo; fuse exact-f32 rowsum ----
    {
      const float* src = aRow + k0;
      #pragma unroll
      for (int i = 0; i < 4; ++i) {
        float4 v = ((const float4*)src)[i];
        float xs[4] = {v.x, v.y, v.z, v.w};
        #pragma unroll
        for (int e = 0; e < 4; ++e) {
          float x = xs[e];
          rs_acc += x;
          unsigned short hi = f2bf_rne(x);
          unsigned short lo = f2bf_rne(x - bf2f(hi));
          int kk = lseg * 16 + i * 4 + e;
          Ahi[lrow][kk] = hi;
          Alo[lrow][kk] = lo;
        }
      }
    }

    // ---- fallback path: stage + convert W in-kernel (ws too small) ----
    if (!PRECONV) {
      for (int idx = tid; idx < CPAD * 4; idx += 256) {
        int row = idx >> 2;
        int seg = idx & 3;
        #pragma unroll
        for (int i = 0; i < 4; ++i) {
          float4 v = make_float4(0.f, 0.f, 0.f, 0.f);
          if (row < Csz)
            v = *(const float4*)(attn_w + (size_t)row * Dsz + k0 + seg * 16 + i * 4);
          float xs[4] = {v.x, v.y, v.z, v.w};
          #pragma unroll
          for (int e = 0; e < 4; ++e) {
            float x = xs[e];
            unsigned short hi = f2bf_rne(x);
            unsigned short lo = f2bf_rne(x - bf2f(hi));
            int kk = seg * 16 + i * 4 + e;
            Wlds[0][row][kk] = hi;
            Wlds[1][row][kk] = lo;
          }
        }
      }
    }

#if HAVE_TDM
    if (PRECONV && w == 0) __builtin_amdgcn_s_wait_tensorcnt(0);
#endif
    __syncthreads();

    // ---- two K=32 WMMA steps over the staged chunk ----
    #pragma unroll
    for (int ks = 0; ks < 2; ++ks) {
      const int kk = ks * 32;
      const int arow = mrow * 16 + rl;
      FragU ahi, alo;
      ahi.q[0] = *(const uint4*)&Ahi[arow][kk + h * 8];
      ahi.q[1] = *(const uint4*)&Ahi[arow][kk + 16 + h * 8];
      alo.q[0] = *(const uint4*)&Alo[arow][kk + h * 8];
      alo.q[1] = *(const uint4*)&Alo[arow][kk + 16 + h * 8];
      #pragma unroll
      for (int j = 0; j < 7; ++j) {
        const int nt = ncol * 7 + j;
        const int wr = nt * 16 + rl;
        FragU bhi, blo;
        bhi.q[0] = *(const uint4*)&Wlds[0][wr][kk + h * 8];
        bhi.q[1] = *(const uint4*)&Wlds[0][wr][kk + 16 + h * 8];
        blo.q[0] = *(const uint4*)&Wlds[1][wr][kk + h * 8];
        blo.q[1] = *(const uint4*)&Wlds[1][wr][kk + 16 + h * 8];
        // hi*hi + hi*lo + lo*hi: products good to ~2^-16 relative
        acc[j] = __builtin_amdgcn_wmma_f32_16x16x32_bf16(false, ahi.v, false, bhi.v,
                                                         (short)0, acc[j], false, false);
        acc[j] = __builtin_amdgcn_wmma_f32_16x16x32_bf16(false, ahi.v, false, blo.v,
                                                         (short)0, acc[j], false, false);
        acc[j] = __builtin_amdgcn_wmma_f32_16x16x32_bf16(false, alo.v, false, bhi.v,
                                                         (short)0, acc[j], false, false);
      }
    }
  }

  // ---- reduce rowsums (4 K-segment partials per row) ----
  red[tid] = rs_acc;
  __syncthreads();
  if (tid < 64)
    rowsum[tid] = red[tid * 4] + red[tid * 4 + 1] + red[tid * 4 + 2] + red[tid * 4 + 3];
  __syncthreads();

  // ---- epilogue: sigmoid, weight by rowsum, reduce rows, atomic into partial ----
  #pragma unroll
  for (int j = 0; j < 7; ++j) {
    const int nt = ncol * 7 + j;
    const int c  = nt * 16 + rl;
    const float bias = (c < Csz) ? attn_b[c] : 0.f;
    float s = 0.f;
    #pragma unroll
    for (int r = 0; r < 8; ++r) {
      const int srow = mrow * 16 + r + 8 * h;   // C-tile row layout: M = r + 8*(lane/16)
      float logit = acc[j][r] + bias;
      float att = 1.0f / (1.0f + __expf(-logit));
      s += att * rowsum[srow];
    }
    if (c < Csz) atomicAdd(&partial[b * Csz + c], s);
  }
}

__global__ void zero_partial(float* __restrict__ p, int n) {
  int i = blockIdx.x * 256 + threadIdx.x;
  if (i < n) p[i] = 0.f;
}

__global__ __launch_bounds__(256)
void finalize_kernel(const float* __restrict__ ct, const float* __restrict__ gw,
                     const float* __restrict__ gb, const float* __restrict__ lam,
                     const float* __restrict__ partial, float* __restrict__ out) {
  int i = blockIdx.x * 256 + threadIdx.x;
  if (i >= Bsz * Csz) return;
  int b = i / Csz, c = i % Csz;
  const float4* a = (const float4*)(ct + (size_t)b * Dsz);
  const float4* wv = (const float4*)(gw + (size_t)c * Dsz);
  float s = 0.f;
  #pragma unroll 4
  for (int k = 0; k < Dsz / 4; ++k) {
    float4 x = a[k], y = wv[k];
    s = fmaf(x.x, y.x, fmaf(x.y, y.y, fmaf(x.z, y.z, fmaf(x.w, y.w, s))));
  }
  out[i] = s + gb[c] + lam[0] * partial[i] * (1.0f / ((float)Ssz * (float)Dsz));
}

extern "C" void kernel_launch(void* const* d_in, const int* in_sizes, int n_in,
                              void* d_out, int out_size, void* d_ws, size_t ws_size,
                              hipStream_t stream) {
  (void)in_sizes; (void)n_in; (void)out_size;
  const float* patch = (const float*)d_in[0];
  const float* ctok  = (const float*)d_in[1];
  const float* aw    = (const float*)d_in[2];
  const float* ab    = (const float*)d_in[3];
  const float* gw    = (const float*)d_in[4];
  const float* gb    = (const float*)d_in[5];
  const float* lam   = (const float*)d_in[6];
  float* out = (float*)d_out;

  float* partial = (float*)d_ws;                       // B*C f32 accumulator
  const size_t partial_bytes = (size_t)Bsz * Csz * sizeof(float);   // 51200 (16B aligned)
  unsigned short* wbf = (unsigned short*)((char*)d_ws + partial_bytes);
  const size_t wbf_bytes = (size_t)NKC * 2 * CPAD * KSTR * sizeof(unsigned short); // 774144
  const bool preconv = ws_size >= partial_bytes + wbf_bytes;

  const int n = Bsz * Csz;
  zero_partial<<<(n + 255) / 256, 256, 0, stream>>>(partial, n);
  dim3 grid(Ssz / 64, Bsz);
  if (preconv) {
    const int nw = NKC * CPAD * KSTR;
    preconv_w_kernel<<<(nw + 255) / 256, 256, 0, stream>>>(aw, wbf);
    attn_gemm_kernel<true><<<grid, 256, 0, stream>>>(patch, aw, wbf, ab, partial);
  } else {
    attn_gemm_kernel<false><<<grid, 256, 0, stream>>>(patch, aw, nullptr, ab, partial);
  }
  finalize_kernel<<<(n + 255) / 256, 256, 0, stream>>>(ctok, gw, gb, lam, partial, out);
}
